// SpectralClusteringGCN_18004502905157
// MI455X (gfx1250) — matrix operand order repaired and checked by reading, not compiled, not run on hardware
//
#include <hip/hip_runtime.h>
#include <hip/hip_bf16.h>
#include <math.h>

typedef __attribute__((ext_vector_type(16))) __bf16 v16bf;
typedef __attribute__((ext_vector_type(8)))  float  v8f;
typedef __attribute__((ext_vector_type(4)))  unsigned int u32x4;
typedef __attribute__((ext_vector_type(8)))  int  i32x8;
typedef __attribute__((ext_vector_type(4)))  int  i32x4;

#define NN      8192
#define EE      262144
#define DIN     1024
#define DOUT    256
#define KCL     8
#define RE      6371.0f

#if __has_builtin(__builtin_amdgcn_tensor_load_to_lds)
#define HAVE_TDM 1
#else
#define HAVE_TDM 0
#endif

#if HAVE_TDM
// Issue a TDM 2D tile load (f32 elements) into LDS. One issue per executing
// wave; EXEC is ignored by tensor ops. Tracked by TENSORcnt.
__device__ __forceinline__ void tdm_load_2d_f32(
    const float* gptr, unsigned lds_off,
    unsigned tensor_w, unsigned tensor_h, unsigned stride0_elems,
    unsigned tile_w, unsigned tile_h)
{
    unsigned long long ga = (unsigned long long)(uintptr_t)gptr;
    u32x4 g0;
    g0[0] = 1u;                                          // count=1, user mode
    g0[1] = lds_off;                                     // lds_addr (bytes)
    g0[2] = (unsigned)(ga & 0xFFFFFFFFu);                // global_addr[31:0]
    g0[3] = (unsigned)((ga >> 32) & 0x1FFFFFFu) | (2u << 30);  // [56:32] | type=2
    i32x8 g1;
    g1[0] = (int)(2u << 16);                             // data_size=2 (4 bytes)
    g1[1] = (int)((tensor_w & 0xFFFFu) << 16);           // tensor_dim0[15:0]
    g1[2] = (int)((tensor_w >> 16) | ((tensor_h & 0xFFFFu) << 16));
    g1[3] = (int)((tensor_h >> 16) | (tile_w << 16));    // tile_dim0
    g1[4] = (int)tile_h;                                 // tile_dim1 (tile_dim2=0)
    g1[5] = (int)stride0_elems;                          // tensor_dim0_stride[31:0]
    g1[6] = 0;
    g1[7] = 0;
    i32x4 z4 = {0, 0, 0, 0};
#if defined(__clang_major__) && __clang_major__ >= 23
    i32x8 z8 = {0, 0, 0, 0, 0, 0, 0, 0};
    __builtin_amdgcn_tensor_load_to_lds(g0, g1, z4, z4, z8, 0);
#else
    __builtin_amdgcn_tensor_load_to_lds(g0, g1, z4, z4, 0);
#endif
}
#endif

// ---------------------------------------------------------------------------
// GEMM: C[M,Nn] = A[M,K] @ W[K,Nn]; fp32 in/out, bf16 WMMA, f32 accumulate.
// WG = 256 threads = 8 waves; tile = 128(M) x 128(N); K chunk = 32.
// A chunk staged by the Tensor Data Mover (double-buffered f32 in LDS);
// W chunk staged by threads (float4 loads -> transposed bf16 in LDS).
// ---------------------------------------------------------------------------
__global__ __launch_bounds__(256) void gemm_bf16_wmma(
    const float* __restrict__ A, const float* __restrict__ W,
    float* __restrict__ C, int M, int K, int Nn)
{
    __shared__ __attribute__((aligned(16))) float  lAf[2][128 * 32];  // 32 KB
    __shared__ __attribute__((aligned(32))) __bf16 lB[128 * 32];      //  8 KB, lB[n*32+k]

    const int tid  = threadIdx.x;
    const int lane = tid & 31;
    const int wave = tid >> 5;
    const int mblk = blockIdx.x * 128;
    const int nblk = blockIdx.y * 128;

    const int row = lane & 15;       // A M-row / B,D N-col within 16x16 tile
    const int hi  = lane >> 4;       // half-wave selector
    const int kbA = hi * 16;         // A frag: lanes 16-31 hold K=16..31
    const int kbB = hi * 16;         // B frag: lanes 16-31 hold K=16..31

    v8f acc[8] = {};

#if HAVE_TDM
    if (wave == 0)
        tdm_load_2d_f32(&A[(size_t)mblk * K + 0],
                        (unsigned)(uintptr_t)&lAf[0][0],
                        (unsigned)K, (unsigned)M, (unsigned)K, 32u, 128u);
#endif

    int buf = 0;
    for (int kc = 0; kc < K; kc += 32) {
        __syncthreads();   // previous compute done: lB and lAf[buf^1] reusable

#if HAVE_TDM
        if (wave == 0 && kc + 32 < K)
            tdm_load_2d_f32(&A[(size_t)mblk * K + (kc + 32)],
                            (unsigned)(uintptr_t)&lAf[buf ^ 1][0],
                            (unsigned)K, (unsigned)M, (unsigned)K, 32u, 128u);
#else
        {   // synchronous vectorized A staging: thread -> 16 contiguous K elems
            int m  = tid >> 1;
            int kh = (tid & 1) * 16;
            const float4* src = (const float4*)&A[(size_t)(mblk + m) * K + kc + kh];
            float4* dst = (float4*)&lAf[0][m * 32 + kh];
            dst[0] = src[0]; dst[1] = src[1]; dst[2] = src[2]; dst[3] = src[3];
        }
#endif

        {   // W staging: thread reads 16 contiguous N at one K row, writes
            // transposed bf16 (coalesced global float4 reads).
            int kk = tid >> 3;
            int n0 = (tid & 7) * 16;
            const float4* wp = (const float4*)&W[(size_t)(kc + kk) * Nn + nblk + n0];
            float4 w0 = wp[0], w1 = wp[1], w2 = wp[2], w3 = wp[3];
            float wv[16] = { w0.x, w0.y, w0.z, w0.w, w1.x, w1.y, w1.z, w1.w,
                             w2.x, w2.y, w2.z, w2.w, w3.x, w3.y, w3.z, w3.w };
            #pragma unroll
            for (int j = 0; j < 16; ++j)
                lB[(n0 + j) * 32 + kk] = (__bf16)wv[j];
        }

#if HAVE_TDM
        if (wave == 0) {
            if (kc + 32 < K) __builtin_amdgcn_s_wait_tensorcnt(1);  // buf ready
            else             __builtin_amdgcn_s_wait_tensorcnt(0);
        }
#endif
        __syncthreads();   // lAf[buf] + lB visible to all waves

        // A fragment: 16x32 bf16: one contiguous 16-elem K run (matches B)
        const float* ap = &lAf[buf][(wave * 16 + row) * 32];
        v16bf afrag;
        #pragma unroll
        for (int e = 0; e < 8; ++e) afrag[e]     = (__bf16)ap[kbA + e];
        #pragma unroll
        for (int e = 0; e < 8; ++e) afrag[8 + e] = (__bf16)ap[kbA + 8 + e];

        // 8 B fragments (K-contiguous in transposed LDS) and 8 WMMAs
        #pragma unroll
        for (int t = 0; t < 8; ++t) {
            v16bf bfrag = *(const v16bf*)&lB[(t * 16 + row) * 32 + kbB];
            acc[t] = __builtin_amdgcn_wmma_f32_16x16x32_bf16(
                false, afrag, false, bfrag, (short)0, acc[t], false, false);
        }
#if HAVE_TDM
        buf ^= 1;
#endif
    }

    // Store D (16x16 f32 layout: VGPR v -> M = v + 8*hi, N = row)
    #pragma unroll
    for (int t = 0; t < 8; ++t) {
        #pragma unroll
        for (int v = 0; v < 8; ++v) {
            int r = mblk + wave * 16 + v + hi * 8;
            int c = nblk + t * 16 + row;
            C[(size_t)r * Nn + c] = acc[t][v];
        }
    }
}

// ---------------------------------------------------------------------------
// Laplacian
// ---------------------------------------------------------------------------
__global__ void node_trig(const float* __restrict__ coords,
                          float* lat, float* lon, float* coslat, int n)
{
    int i = blockIdx.x * blockDim.x + threadIdx.x;
    if (i >= n) return;
    float la = coords[2 * i]     * 0.017453292519943295f;
    float lo = coords[2 * i + 1] * 0.017453292519943295f;
    lat[i] = la; lon[i] = lo; coslat[i] = cosf(la);
}

__global__ __launch_bounds__(256) void lap_pass1(
    const float* __restrict__ lat, const float* __restrict__ lon,
    const float* __restrict__ coslat, float* __restrict__ sim_out,
    float* __restrict__ deg_geo, int n)
{
    __shared__ float red[256];
    const int i = blockIdx.x;
    const float lati = lat[i], loni = lon[i], ci = coslat[i];
    float s = 0.f;
    for (int j = threadIdx.x; j < n; j += 256) {
        float v;
        if (j == i) {
            v = 0.f;
        } else {
            float sdlat = sinf(0.5f * (lati - lat[j]));
            float sdlon = sinf(0.5f * (loni - lon[j]));
            float a = sdlat * sdlat + ci * coslat[j] * sdlon * sdlon;
            a = fminf(fmaxf(a, 0.f), 1.f);
            v = 2.f * RE * asinf(sqrtf(a));
        }
        sim_out[(size_t)i * n + j] = v;
        s += v;
    }
    red[threadIdx.x] = s;
    __syncthreads();
    for (int off = 128; off > 0; off >>= 1) {
        if (threadIdx.x < off) red[threadIdx.x] += red[threadIdx.x + off];
        __syncthreads();
    }
    if (threadIdx.x == 0) deg_geo[i] = red[0];
}

__global__ void rsqrt_vec(const float* __restrict__ deg, float* __restrict__ dinv,
                          int n, float addend)
{
    int i = blockIdx.x * blockDim.x + threadIdx.x;
    if (i < n) dinv[i] = rsqrtf(deg[i] + addend);
}

__global__ void lap_pass2(float* __restrict__ lap, const float* __restrict__ dinv, int n)
{
    size_t idx = (size_t)blockIdx.x * blockDim.x + threadIdx.x;
    if (idx >= (size_t)n * n) return;
    int i = (int)(idx / n), j = (int)(idx % n);
    lap[idx] = (i == j) ? 1.0f : -dinv[i] * lap[idx] * dinv[j];
}

// ---------------------------------------------------------------------------
// GCN aggregation
// ---------------------------------------------------------------------------
__global__ void zero_f(float* p, size_t n)
{
    size_t i = (size_t)blockIdx.x * blockDim.x + threadIdx.x;
    if (i < n) p[i] = 0.f;
}

__global__ void edge_deg(const long long* __restrict__ dst, float* deg, int e)
{
    int i = blockIdx.x * blockDim.x + threadIdx.x;
    if (i < e) atomicAdd(&deg[(int)dst[i]], 1.0f);
}

__global__ void agg_self(const float* __restrict__ h, const float* __restrict__ dinv,
                         float* __restrict__ agg, size_t total, int D)
{
    size_t idx = (size_t)blockIdx.x * blockDim.x + threadIdx.x;
    if (idx >= total) return;
    int i = (int)(idx / D);
    float dv = dinv[i];
    agg[idx] = h[idx] * dv * dv;
}

__global__ void edge_scatter(const float* __restrict__ h, const float* __restrict__ dinv,
                             const long long* __restrict__ src,
                             const long long* __restrict__ dst,
                             float* __restrict__ agg, int e, int D)
{
    const int q = D >> 2;                         // float4 slices per edge
    size_t idx = (size_t)blockIdx.x * blockDim.x + threadIdx.x;
    if (idx >= (size_t)e * q) return;
    int ei = (int)(idx / q);
    int d4 = (int)(idx % q) * 4;
    int s = (int)src[ei], d = (int)dst[ei];
    float w = dinv[s] * dinv[d];
    const float4 hv = *(const float4*)&h[(size_t)s * D + d4];
    float* ap = &agg[(size_t)d * D + d4];
    atomicAdd(ap + 0, hv.x * w);
    atomicAdd(ap + 1, hv.y * w);
    atomicAdd(ap + 2, hv.z * w);
    atomicAdd(ap + 3, hv.w * w);
}

__global__ void bias_act(const float* __restrict__ agg, const float* __restrict__ b,
                         float* __restrict__ out, size_t total, int D, int relu)
{
    size_t idx = (size_t)blockIdx.x * blockDim.x + threadIdx.x;
    if (idx >= total) return;
    float v = agg[idx] + b[idx % D];
    out[idx] = relu ? fmaxf(v, 0.f) : v;
}

// ---------------------------------------------------------------------------
// K-means (K=8, D=256)
// ---------------------------------------------------------------------------
__global__ void km_init(const float* __restrict__ h, float* cents)
{
    int i = blockIdx.x * blockDim.x + threadIdx.x;
    if (i < KCL * DOUT) cents[i] = h[i];
}

__global__ __launch_bounds__(256) void km_assign(
    const float* __restrict__ x, const float* __restrict__ cents,
    float* csum, float* ccnt, float* labels_out, int n, int accum)
{
    __shared__ float lc[KCL * DOUT];
    for (int i = threadIdx.x; i < KCL * DOUT; i += 256) lc[i] = cents[i];
    __syncthreads();
    int i = blockIdx.x * 256 + threadIdx.x;
    if (i >= n) return;
    const float* xi = &x[(size_t)i * DOUT];
    float best = 3.4e38f; int bk = 0;
    #pragma unroll
    for (int k = 0; k < KCL; ++k) {
        const float* ck = &lc[k * DOUT];
        float d = 0.f;
        for (int t = 0; t < DOUT; ++t) { float df = xi[t] - ck[t]; d = fmaf(df, df, d); }
        if (d < best) { best = d; bk = k; }
    }
    if (accum) {
        atomicAdd(&ccnt[bk], 1.0f);
        for (int t = 0; t < DOUT; ++t) atomicAdd(&csum[bk * DOUT + t], xi[t]);
    }
    if (labels_out) labels_out[i] = (float)bk;
}

__global__ void km_update(float* cents, const float* __restrict__ csum,
                          const float* __restrict__ ccnt)
{
    int idx = blockIdx.x * blockDim.x + threadIdx.x;
    if (idx >= KCL * DOUT) return;
    float c = ccnt[idx >> 8];
    if (c > 0.f) cents[idx] = csum[idx] / fmaxf(c, 1.f);
}

// ---------------------------------------------------------------------------
// Host
// ---------------------------------------------------------------------------
extern "C" void kernel_launch(void* const* d_in, const int* in_sizes, int n_in,
                              void* d_out, int out_size, void* d_ws, size_t ws_size,
                              hipStream_t stream)
{
    const float*     x     = (const float*)d_in[0];
    const float*     crd   = (const float*)d_in[1];
    const long long* edge  = (const long long*)d_in[2];   // int64 (2,E)
    const float*     W0    = (const float*)d_in[3];
    const float*     b0    = (const float*)d_in[4];
    const float*     W1    = (const float*)d_in[5];
    const float*     b1    = (const float*)d_in[6];
    const float*     W2    = (const float*)d_in[7];
    const float*     b2    = (const float*)d_in[8];
    const long long* esrc  = edge;
    const long long* edst  = edge + EE;

    float* out     = (float*)d_out;
    float* out_h   = out;                              // N x 256
    float* out_lbl = out + (size_t)NN * DOUT;          // N
    float* out_lap = out_lbl + NN;                     // N x N

    // Workspace layout
    char* w = (char*)d_ws;
    float* f0   = (float*)w; w += (size_t)NN * DIN * 4;   // activations
    float* f1   = (float*)w; w += (size_t)NN * DIN * 4;   // h = A @ W
    float* f2   = (float*)w; w += (size_t)NN * DIN * 4;   // aggregated
    float* dgeo = (float*)w; w += NN * 4;
    float* vgeo = (float*)w; w += NN * 4;
    float* dgcn = (float*)w; w += NN * 4;
    float* vgcn = (float*)w; w += NN * 4;
    float* lat  = (float*)w; w += NN * 4;
    float* lon  = (float*)w; w += NN * 4;
    float* clat = (float*)w; w += NN * 4;
    float* cent = (float*)w; w += KCL * DOUT * 4;
    float* csum = (float*)w; w += KCL * DOUT * 4;
    float* ccnt = (float*)w; w += KCL * 4;

    // ---- Laplacian ----
    node_trig<<<NN / 256, 256, 0, stream>>>(crd, lat, lon, clat, NN);
    lap_pass1<<<NN, 256, 0, stream>>>(lat, lon, clat, out_lap, dgeo, NN);
    rsqrt_vec<<<NN / 256, 256, 0, stream>>>(dgeo, vgeo, NN, 0.f);
    lap_pass2<<<(unsigned)(((size_t)NN * NN) / 256), 256, 0, stream>>>(out_lap, vgeo, NN);

    // ---- Graph degree (shared by all layers) ----
    zero_f<<<NN / 256, 256, 0, stream>>>(dgcn, NN);
    edge_deg<<<EE / 256, 256, 0, stream>>>(edst, dgcn, EE);
    rsqrt_vec<<<NN / 256, 256, 0, stream>>>(dgcn, vgcn, NN, 1.f);

    // ---- GCN layers ----
    const float* Ain = x;
    const float* Ws[3] = {W0, W1, W2};
    const float* bs[3] = {b0, b1, b2};
    const int    Dd[3] = {DIN, DIN, DOUT};
    for (int L = 0; L < 3; ++L) {
        const int D = Dd[L];
        const size_t tot = (size_t)NN * D;
        dim3 gg(NN / 128, D / 128);
        gemm_bf16_wmma<<<gg, 256, 0, stream>>>(Ain, Ws[L], f1, NN, DIN, D);
        agg_self<<<(unsigned)(tot / 256), 256, 0, stream>>>(f1, vgcn, f2, tot, D);
        edge_scatter<<<(unsigned)(((size_t)EE * (D >> 2)) / 256), 256, 0, stream>>>(
            f1, vgcn, esrc, edst, f2, EE, D);
        float* act = (L == 2) ? out_h : f0;
        bias_act<<<(unsigned)(tot / 256), 256, 0, stream>>>(f2, bs[L], act, tot, D, L < 2);
        Ain = f0;
    }

    // ---- K-means on out_h ----
    km_init<<<KCL, DOUT, 0, stream>>>(out_h, cent);
    for (int it = 0; it < 10; ++it) {
        zero_f<<<KCL, DOUT, 0, stream>>>(csum, (size_t)KCL * DOUT);
        zero_f<<<1, 32, 0, stream>>>(ccnt, KCL);
        km_assign<<<NN / 256, 256, 0, stream>>>(out_h, cent, csum, ccnt, nullptr, NN, 1);
        km_update<<<KCL, DOUT, 0, stream>>>(cent, csum, ccnt);
    }
    km_assign<<<NN / 256, 256, 0, stream>>>(out_h, cent, nullptr, nullptr, out_lbl, NN, 0);
}
